// GAT_85504208929185
// MI455X (gfx1250) — compile-verified, matrix operand order checked by reading, not run on hardware
//
#include <hip/hip_runtime.h>

#define NN 100000
#define EE 1600000
#define ENTOT (EE + NN)
#define DH 128
#define NLAYERS 3
#define WPB 8   // waves per 256-thread block (wave32)

typedef __attribute__((ext_vector_type(16))) __bf16 v16bf;
typedef __attribute__((ext_vector_type(8)))  __bf16 v8bf;
typedef __attribute__((ext_vector_type(8)))  float  v8f;

// ---------- helpers ----------
__device__ __forceinline__ unsigned short f2bf(float f) {
  unsigned u = __float_as_uint(f);
  u += 0x7FFFu + ((u >> 16) & 1u);      // round-to-nearest-even
  return (unsigned short)(u >> 16);
}
// order-preserving float<->uint encode for atomicMax on signed floats
__device__ __forceinline__ unsigned fenc(float f) {
  unsigned u = __float_as_uint(f);
  return (u & 0x80000000u) ? ~u : (u | 0x80000000u);
}
__device__ __forceinline__ float fdec(unsigned u) {
  unsigned b = (u & 0x80000000u) ? (u & 0x7FFFFFFFu) : ~u;
  return __uint_as_float(b);
}
__device__ __forceinline__ float wred(float v) {   // wave32 sum-reduce
  #pragma unroll
  for (int o = 16; o > 0; o >>= 1) v += __shfl_xor(v, o, 32);
  return v;
}

// ---------- weight convert: Wt[m][n][k] = W_m[k][n] as bf16 ----------
__global__ __launch_bounds__(256) void k_conv_weights(
    const float* __restrict__ encW, const float* __restrict__ Wg,
    unsigned short* __restrict__ Wt) {
  int id  = blockIdx.x * 256 + threadIdx.x;        // 4*128*128 total
  int k   = id & (DH - 1);
  int n   = (id >> 7) & (DH - 1);
  int mat = id >> 14;
  float v = (mat == 0) ? encW[k * DH + n]
                       : Wg[(size_t)(mat - 1) * DH * DH + k * DH + n];
  Wt[id] = f2bf(v);
}

// ---------- f32 -> bf16, 4-wide (only needed for input x) ----------
__global__ __launch_bounds__(256) void k_to_bf16(
    const float* __restrict__ src, unsigned short* __restrict__ dst, int n4) {
  int i = blockIdx.x * 256 + threadIdx.x;
  if (i >= n4) return;
  float4 v = ((const float4*)src)[i];
  ushort4 o;
  o.x = f2bf(v.x); o.y = f2bf(v.y); o.z = f2bf(v.z); o.w = f2bf(v.w);
  ((ushort4*)dst)[i] = o;
}

// ---------- GEMM: C[M,128] = A[M,128](bf16) * W, W given transposed bf16 ----------
// one wave per 16x16 tile; block = 8 waves = one 16-row stripe x 128 cols; K=128 -> 4 WMMAs
// Optionally also emits a bf16 copy of C (fused conversion for the next GEMM).
// M must be a multiple of 16 (NN = 100000 = 6250*16).
__global__ __launch_bounds__(256) void k_gemm_wmma(
    const unsigned short* __restrict__ A, const unsigned short* __restrict__ Bt,
    const float* __restrict__ bias, float* __restrict__ C,
    unsigned short* __restrict__ Cb) {
  const int lane  = threadIdx.x & 31;
  const int ntile = threadIdx.x >> 5;                 // 0..7 -> 16-col tile
  const int row   = blockIdx.x * 16 + (lane & 15);    // A row for this lane
  const int col   = ntile * 16 + (lane & 15);         // global column
  const int kb    = (lane >> 4) << 3;                 // 0 or 8 (ISA A/B frag layout)
  const unsigned short* arow = A  + (size_t)row * DH;
  const unsigned short* brow = Bt + (size_t)col * DH;
  union U { v16bf v; v8bf h[2]; };
  v8f acc = {};
  #pragma unroll
  for (int ks = 0; ks < 4; ++ks) {
    const int k0 = ks * 32 + kb;
    U a, b;
    a.h[0] = *(const v8bf*)(arow + k0);
    a.h[1] = *(const v8bf*)(arow + k0 + 16);
    b.h[0] = *(const v8bf*)(brow + k0);
    b.h[1] = *(const v8bf*)(brow + k0 + 16);
    acc = __builtin_amdgcn_wmma_f32_16x16x32_bf16(
        false, a.v, false, b.v, (short)0, acc, false, false);
  }
  const int rbase = blockIdx.x * 16 + ((lane >> 4) << 3);
  const float bv = bias ? bias[col] : 0.0f;
  #pragma unroll
  for (int r = 0; r < 8; ++r) {
    float v = acc[r] + bv;
    C[(size_t)(rbase + r) * DH + col] = v;
    if (Cb) Cb[(size_t)(rbase + r) * DH + col] = f2bf(v);
  }
}

// ---------- per-node attention dots: as[n]=hW[n].a_s, ad[n]=hW[n].a_d ----------
__global__ __launch_bounds__(256) void k_attdot(
    const float* __restrict__ hW, const float* __restrict__ avs,
    const float* __restrict__ avd, float* __restrict__ As,
    float* __restrict__ Ad, int n) {
  int wid  = blockIdx.x * WPB + (threadIdx.x >> 5);
  int lane = threadIdx.x & 31;
  if (wid >= n) return;
  float4 h4 = ((const float4*)(hW + (size_t)wid * DH))[lane];
  float4 s4 = ((const float4*)avs)[lane];
  float4 d4 = ((const float4*)avd)[lane];
  float ps = h4.x*s4.x + h4.y*s4.y + h4.z*s4.z + h4.w*s4.w;
  float pd = h4.x*d4.x + h4.y*d4.y + h4.z*d4.z + h4.w*d4.w;
  ps = wred(ps); pd = wred(pd);
  if (lane == 0) { As[wid] = ps; Ad[wid] = pd; }
}

// ---------- init per-layer segment state ----------
__global__ __launch_bounds__(256) void k_init_seg(
    unsigned* __restrict__ mkey, float* __restrict__ den, int n) {
  int i = blockIdx.x * 256 + threadIdx.x;
  if (i >= n) return;
  mkey[i] = 0x007FFFFFu;   // fenc(-inf)
  den[i]  = 0.0f;
}

// ---------- per-edge score + segment max ----------
__global__ __launch_bounds__(256) void k_edge_score(
    const int* __restrict__ src, const int* __restrict__ dst,
    const float* __restrict__ As, const float* __restrict__ Ad,
    float* __restrict__ e, unsigned* __restrict__ mkey) {
  int i = blockIdx.x * 256 + threadIdx.x;
  if (i >= ENTOT) return;
  int s = (i < EE) ? src[i] : (i - EE);   // self-loops appended
  int d = (i < EE) ? dst[i] : (i - EE);
  float v = As[s] + Ad[d];
  v = (v > 0.0f) ? v : 0.2f * v;          // leaky_relu
  e[i] = v;
  atomicMax(mkey + d, fenc(v));
}

// ---------- per-edge aggregation: acc[dst] += exp(e-m)*hW[src]; den[dst] += exp ----------
__global__ __launch_bounds__(256) void k_edge_agg(
    const int* __restrict__ src, const int* __restrict__ dst,
    const float* __restrict__ e, const unsigned* __restrict__ mkey,
    const float* __restrict__ hW, float* __restrict__ acc,
    float* __restrict__ den) {
  int wid  = blockIdx.x * WPB + (threadIdx.x >> 5);
  int lane = threadIdx.x & 31;
  if (wid >= ENTOT) return;
  int s = (wid < EE) ? src[wid] : (wid - EE);
  int d = (wid < EE) ? dst[wid] : (wid - EE);
  float ex = __expf(e[wid] - fdec(mkey[d]));
  if (lane == 0) atomicAdd(den + d, ex);
  float4 v = ((const float4*)(hW + (size_t)s * DH))[lane];
  float* o = acc + (size_t)d * DH + lane * 4;
  atomicAdd(o + 0, ex * v.x);
  atomicAdd(o + 1, ex * v.y);
  atomicAdd(o + 2, ex * v.z);
  atomicAdd(o + 3, ex * v.w);
}

// ---------- per-node normalize + bias, accumulate LN stats ----------
__global__ __launch_bounds__(256) void k_node_fin(
    float* __restrict__ acc, const float* __restrict__ den,
    const float* __restrict__ bgl, float* __restrict__ stats, int n) {
  int wid  = blockIdx.x * WPB + (threadIdx.x >> 5);
  int lane = threadIdx.x & 31;
  if (wid >= n) return;
  float inv = 1.0f / den[wid];
  float4* a4 = (float4*)(acc + (size_t)wid * DH);
  float4 b4 = ((const float4*)bgl)[lane];
  float4 y = a4[lane];
  y.x = y.x * inv + b4.x; y.y = y.y * inv + b4.y;
  y.z = y.z * inv + b4.z; y.w = y.w * inv + b4.w;
  a4[lane] = y;
  float s = y.x + y.y + y.z + y.w;
  float q = y.x*y.x + y.y*y.y + y.z*y.z + y.w*y.w;
  s = wred(s); q = wred(q);
  if (lane == 0) { atomicAdd(stats, s); atomicAdd(stats + 1, q); }
}

// ---------- graph-LN + relu + residual: h = relu(ln(y)) + h ; also emit bf16(h) ----------
__global__ __launch_bounds__(256) void k_apply_ln(
    float* __restrict__ h, unsigned short* __restrict__ hb,
    const float* __restrict__ y, const float* __restrict__ lnw,
    const float* __restrict__ lnb, const float* __restrict__ stats, int n4) {
  int i = blockIdx.x * 256 + threadIdx.x;
  if (i >= n4) return;
  const float cnt  = (float)NN * (float)DH;
  float mean = stats[0] / cnt;
  float var  = stats[1] / cnt - mean * mean;
  float inv  = rsqrtf(var + 1e-5f);
  int c4 = i & 31;
  float4 yv = ((const float4*)y)[i];
  float4 w  = ((const float4*)lnw)[c4];
  float4 b  = ((const float4*)lnb)[c4];
  float4 hp = ((float4*)h)[i];
  float4 t;
  t.x = fmaxf(w.x * (yv.x - mean) * inv + b.x, 0.0f) + hp.x;
  t.y = fmaxf(w.y * (yv.y - mean) * inv + b.y, 0.0f) + hp.y;
  t.z = fmaxf(w.z * (yv.z - mean) * inv + b.z, 0.0f) + hp.z;
  t.w = fmaxf(w.w * (yv.w - mean) * inv + b.w, 0.0f) + hp.w;
  ((float4*)h)[i] = t;
  ushort4 o;
  o.x = f2bf(t.x); o.y = f2bf(t.y); o.z = f2bf(t.z); o.w = f2bf(t.w);
  ((ushort4*)hb)[i] = o;
}

// ---------- decoder: out[0] = sum_n sigmoid(h[n].decW + decb) ----------
__global__ __launch_bounds__(256) void k_decoder(
    const float* __restrict__ h, const float* __restrict__ decW,
    const float* __restrict__ decb, float* __restrict__ out, int n) {
  int wid  = blockIdx.x * WPB + (threadIdx.x >> 5);
  int lane = threadIdx.x & 31;
  if (wid >= n) return;
  float4 hv = ((const float4*)(h + (size_t)wid * DH))[lane];
  float4 w  = ((const float4*)decW)[lane];
  float p = hv.x*w.x + hv.y*w.y + hv.z*w.z + hv.w*w.w;
  p = wred(p);
  if (lane == 0) {
    float z = 1.0f / (1.0f + __expf(-(p + decb[0])));
    atomicAdd(out, z);
  }
}

extern "C" void kernel_launch(void* const* d_in, const int* in_sizes, int n_in,
                              void* d_out, int out_size, void* d_ws, size_t ws_size,
                              hipStream_t stream) {
  const float* x     = (const float*)d_in[0];
  const int*   ei    = (const int*)d_in[1];
  const float* encW  = (const float*)d_in[2];
  const float* encb  = (const float*)d_in[3];
  const float* Wg    = (const float*)d_in[4];
  const float* a_src = (const float*)d_in[5];
  const float* a_dst = (const float*)d_in[6];
  const float* bg    = (const float*)d_in[7];
  const float* lnw   = (const float*)d_in[8];
  const float* lnb   = (const float*)d_in[9];
  const float* decW  = (const float*)d_in[10];
  const float* decb  = (const float*)d_in[11];
  const int* src = ei;
  const int* dst = ei + EE;

  char* p = (char*)d_ws;
  auto carve = [&](size_t bytes) {
    char* r = p; p += (bytes + 255) & ~(size_t)255; return (void*)r;
  };
  float*          bufH   = (float*)carve((size_t)NN * DH * 4);
  unsigned short* bufHb  = (unsigned short*)carve((size_t)NN * DH * 2);
  unsigned short* bufXb  = (unsigned short*)carve((size_t)NN * DH * 2);
  float*          bufHW  = (float*)carve((size_t)NN * DH * 4);
  float*          bufAcc = (float*)carve((size_t)NN * DH * 4);
  float*          bufAs  = (float*)carve((size_t)NN * 4);
  float*          bufAd  = (float*)carve((size_t)NN * 4);
  unsigned*       bufM   = (unsigned*)carve((size_t)NN * 4);
  float*          bufDen = (float*)carve((size_t)NN * 4);
  float*          bufE   = (float*)carve((size_t)ENTOT * 4);
  unsigned short* bufWt  = (unsigned short*)carve((size_t)4 * DH * DH * 2);
  float*          stats  = (float*)carve(32);

  const int nodeWaveBlocks = (NN + WPB - 1) / WPB;          // 12500
  const int edgeWaveBlocks = (ENTOT + WPB - 1) / WPB;       // 212500
  const int elem4Blocks    = (NN * (DH / 4) + 255) / 256;   // 12500

  // weights -> transposed bf16 (enc + 3 layers)
  k_conv_weights<<<(4 * DH * DH) / 256, 256, 0, stream>>>(encW, Wg, bufWt);

  // encoder: h = x @ enc_W + enc_b ; also emits bf16(h) for layer 0's GEMM
  k_to_bf16<<<elem4Blocks, 256, 0, stream>>>(x, bufXb, NN * (DH / 4));
  k_gemm_wmma<<<NN / 16, 256, 0, stream>>>(bufXb, bufWt, encb, bufH, bufHb);

  for (int l = 0; l < NLAYERS; ++l) {
    // hW = h @ Wg[l]   (A = bf16(h) produced by the previous stage)
    k_gemm_wmma<<<NN / 16, 256, 0, stream>>>(
        bufHb, bufWt + (size_t)(1 + l) * DH * DH, nullptr, bufHW, nullptr);
    // attention logits per node
    k_attdot<<<nodeWaveBlocks, 256, 0, stream>>>(
        bufHW, a_src + l * DH, a_dst + l * DH, bufAs, bufAd, NN);
    // reset segment state
    k_init_seg<<<(NN + 255) / 256, 256, 0, stream>>>(bufM, bufDen, NN);
    hipMemsetAsync(bufAcc, 0, (size_t)NN * DH * 4, stream);
    hipMemsetAsync(stats, 0, 8, stream);
    // softmax: max, then unnormalized accumulate, then divide per node
    k_edge_score<<<(ENTOT + 255) / 256, 256, 0, stream>>>(
        src, dst, bufAs, bufAd, bufE, bufM);
    k_edge_agg<<<edgeWaveBlocks, 256, 0, stream>>>(
        src, dst, bufE, bufM, bufHW, bufAcc, bufDen);
    k_node_fin<<<nodeWaveBlocks, 256, 0, stream>>>(
        bufAcc, bufDen, bg + l * DH, stats, NN);
    // graph layernorm + relu + residual into h ; fused bf16(h) for next layer
    k_apply_ln<<<elem4Blocks, 256, 0, stream>>>(
        bufH, bufHb, bufAcc, lnw + l * DH, lnb + l * DH, stats, NN * (DH / 4));
  }

  hipMemsetAsync(d_out, 0, (size_t)out_size * 4, stream);
  k_decoder<<<nodeWaveBlocks, 256, 0, stream>>>(bufH, decW, decb, (float*)d_out, NN);
  (void)in_sizes; (void)n_in; (void)ws_size;
}
